// PlainMultiheadAttentionSingLoRA_28355374088385
// MI455X (gfx1250) — compile-verified, matrix operand order checked by reading, not run on hardware
//
#include <hip/hip_runtime.h>
#include <math.h>

typedef __attribute__((ext_vector_type(16))) _Float16 v16h;
typedef __attribute__((ext_vector_type(8)))  float    v8f;

#define D_ 768
#define B_ 2
#define S_ 2048
#define H_ 12
#define HD_ 64
#define R_ 16
#define M_ (B_ * S_)            // 4096 rows
#define LORA_SCALE 4.0f         // alpha / sqrt(R) = 16/4
#define ATTN_SCALE 0.125f       // 1/sqrt(64)

union TileHalf {
  v16h v;
  _Float16 h[16];
  float4 f4[2];
};

// Load a 16x32 f16 tile (row-major, stride in halves) into the WMMA 16-bit
// A-operand VGPR layout (also used for B-operand as B^T):
//   lane: row = lane&15, hi = lane>>4
//   halves 0..7  hold K = 8*hi + 0..7
//   halves 8..15 hold K = 16 + 8*hi + 0..7
__device__ __forceinline__ v16h load_tile_a(const _Float16* __restrict__ base,
                                            int stride_halves) {
  const int lane = threadIdx.x & 31;
  const int r = lane & 15;
  const int hi = lane >> 4;
  const _Float16* p = base + (size_t)r * stride_halves + 8 * hi;
  TileHalf t;
  t.f4[0] = *reinterpret_cast<const float4*>(p);        // K = 8*hi .. 8*hi+7
  t.f4[1] = *reinterpret_cast<const float4*>(p + 16);   // K = 16+8*hi .. +7
  return t.v;
}

__device__ __forceinline__ v8f wmma_f16(v16h a, v16h b, v8f c) {
  return __builtin_amdgcn_wmma_f32_16x16x32_f16(false, a, false, b,
                                                (short)0, c, false, false);
}

__device__ __forceinline__ v8f vzero8() {
  v8f z = {};
  return z;
}

// ---------------------------------------------------------------- prep kernels
__global__ void k_f32_to_f16(const float* __restrict__ in,
                             _Float16* __restrict__ out, int n) {
  int i = blockIdx.x * blockDim.x + threadIdx.x;
  if (i < n) out[i] = (_Float16)in[i];
}

// Wh[n][k] = W[n][k] + 4 * (A A^T)[n][k]   (this is W_eff^T, the B-operand)
__global__ void k_build_weight(const float* __restrict__ W,
                               const float* __restrict__ A,
                               _Float16* __restrict__ Wh) {
  int i = blockIdx.x * blockDim.x + threadIdx.x;
  if (i >= D_ * D_) return;
  int n = i / D_, k = i % D_;
  float acc = 0.f;
#pragma unroll
  for (int r = 0; r < R_; ++r) acc += A[n * R_ + r] * A[k * R_ + r];
  Wh[i] = (_Float16)(W[i] + LORA_SCALE * acc);
}

// ---------------------------------------------------------------- WMMA GEMM
// Y[m][n] = bias[n] + sum_k Ah[m][k] * Bh[n][k]
// Wave tile: 32(M) x 64(N). K-loop unrolled x2 with explicit ping-pong
// register buffers so the double-buffer rotation costs no v_mov copies.
// mode 0: store f16 row-major [M_, D_]
// mode 1: store f16 per-head transposed: vt[((b*H+h)*64+d)*S + s]
// mode 2: store f32 row-major [M_, D_] (final output)
__global__ void __launch_bounds__(128)
k_gemm(const _Float16* __restrict__ Ah, const _Float16* __restrict__ Bh,
       const float* __restrict__ bias, _Float16* __restrict__ outh,
       float* __restrict__ outf, int mode) {
  const int wave = (blockIdx.x * blockDim.x + threadIdx.x) >> 5;
  const int tilesN = D_ / 64;  // 12
  const int mrow = (wave / tilesN) * 32;
  const int ncol = (wave % tilesN) * 64;
  if (mrow >= M_) return;
  const int lane = threadIdx.x & 31;
  const int r = lane & 15, hi = lane >> 4;

  v8f acc[2][4];
#pragma unroll
  for (int mi = 0; mi < 2; ++mi)
#pragma unroll
    for (int nt = 0; nt < 4; ++nt) acc[mi][nt] = vzero8();

  const _Float16* arow0 = Ah + (size_t)mrow * D_;
  const _Float16* arow1 = Ah + (size_t)(mrow + 16) * D_;
  const _Float16* brow[4];
#pragma unroll
  for (int nt = 0; nt < 4; ++nt) brow[nt] = Bh + (size_t)(ncol + nt * 16) * D_;

  // ---- prologue: load set A at k = 0
  v16h a0A = load_tile_a(arow0, D_);
  v16h a1A = load_tile_a(arow1, D_);
  v16h bA[4];
#pragma unroll
  for (int nt = 0; nt < 4; ++nt) bA[nt] = load_tile_a(brow[nt], D_);

  // D_/32 = 24 k-steps, processed 2 per iteration (12 iterations)
  for (int k0 = 0; k0 < D_; k0 += 64) {
    const int k1 = k0 + 32;                    // always < D_ (D_ % 64 == 0)
    int k2 = k0 + 64;
    if (k2 > D_ - 32) k2 = D_ - 32;            // clamp: redundant last reload

    // load set B @ k1 (overlaps the 8 WMMAs on set A)
    v16h a0B = load_tile_a(arow0 + k1, D_);
    v16h a1B = load_tile_a(arow1 + k1, D_);
    v16h bB[4];
#pragma unroll
    for (int nt = 0; nt < 4; ++nt) bB[nt] = load_tile_a(brow[nt] + k1, D_);

#pragma unroll
    for (int nt = 0; nt < 4; ++nt) {
      acc[0][nt] = wmma_f16(a0A, bA[nt], acc[0][nt]);
      acc[1][nt] = wmma_f16(a1A, bA[nt], acc[1][nt]);
    }

    // load set A @ k2 (overlaps the 8 WMMAs on set B)
    a0A = load_tile_a(arow0 + k2, D_);
    a1A = load_tile_a(arow1 + k2, D_);
#pragma unroll
    for (int nt = 0; nt < 4; ++nt) bA[nt] = load_tile_a(brow[nt] + k2, D_);

#pragma unroll
    for (int nt = 0; nt < 4; ++nt) {
      acc[0][nt] = wmma_f16(a0B, bB[nt], acc[0][nt]);
      acc[1][nt] = wmma_f16(a1B, bB[nt], acc[1][nt]);
    }
  }

#pragma unroll
  for (int mi = 0; mi < 2; ++mi) {
#pragma unroll
    for (int nt = 0; nt < 4; ++nt) {
#pragma unroll
      for (int v = 0; v < 8; ++v) {
        int row = mrow + mi * 16 + v + 8 * hi;  // C layout: row = v + 8*hi
        int col = ncol + nt * 16 + r;           //           col = lane & 15
        float val = acc[mi][nt][v] + bias[col];
        if (mode == 0) {
          outh[(size_t)row * D_ + col] = (_Float16)val;
        } else if (mode == 1) {
          int bb = row / S_, s = row % S_;
          int h = col / HD_, d = col % HD_;
          outh[((size_t)(bb * H_ + h) * HD_ + d) * S_ + s] = (_Float16)val;
        } else {
          outf[(size_t)row * D_ + col] = val;
        }
      }
    }
  }
}

// ---------------------------------------------------------------- attention
// One wave handles one (b, h, 16-query block); streams keys in blocks of 32
// with an online (flash) softmax. V tiles are loaded up-front each iteration
// so the softmax VALU work and LDS P bounce overlap their latency.
__global__ void __launch_bounds__(128)
k_attn(const _Float16* __restrict__ qh, const _Float16* __restrict__ kh,
       const _Float16* __restrict__ vt, _Float16* __restrict__ aoh) {
  __shared__ _Float16 pbuf[4][16 * 32];   // per-wave P bounce buffer (1 KB)
  const int wave = (blockIdx.x * blockDim.x + threadIdx.x) >> 5;
  const int wslot = (threadIdx.x >> 5) & 3;
  const int lane = threadIdx.x & 31;
  const int r = lane & 15, hi = lane >> 4;
  const int NQ = S_ / 16;  // 128 query blocks per head

  int idx = wave;
  const int qb = idx % NQ; idx /= NQ;
  const int h  = idx % H_; idx /= H_;
  const int b  = idx;
  if (b >= B_) return;

  // Q tile (16 x 64) held in registers as two A-operand tiles (K=32 each)
  const _Float16* qbase = qh + ((size_t)(b * S_) + qb * 16) * D_ + h * HD_;
  v16h a0 = load_tile_a(qbase, D_);
  v16h a1 = load_tile_a(qbase + 32, D_);

  v8f o[4];
#pragma unroll
  for (int dt = 0; dt < 4; ++dt) o[dt] = vzero8();
  float mi[8], li[8];
#pragma unroll
  for (int v = 0; v < 8; ++v) { mi[v] = -1e30f; li[v] = 0.f; }

  const _Float16* vhead = vt + (size_t)(b * H_ + h) * HD_ * S_;

  for (int kb = 0; kb < S_; kb += 32) {
    const _Float16* kb0 = kh + ((size_t)(b * S_) + kb) * D_ + h * HD_;

    // ---- issue ALL loads for this key block up front (K then V tiles)
    v16h bk[2][2];
#pragma unroll
    for (int nt = 0; nt < 2; ++nt) {
      bk[nt][0] = load_tile_a(kb0 + (size_t)(nt * 16) * D_, D_);
      bk[nt][1] = load_tile_a(kb0 + (size_t)(nt * 16) * D_ + 32, D_);
    }
    v16h bv[4];
#pragma unroll
    for (int dt = 0; dt < 4; ++dt)
      bv[dt] = load_tile_a(vhead + (size_t)(dt * 16) * S_ + kb, S_);

    // ---- scores S[16 x 32] = Q @ K^T / 8 : two 16x16 C tiles
    v8f sc[2];
    sc[0] = vzero8();
    sc[1] = vzero8();
#pragma unroll
    for (int nt = 0; nt < 2; ++nt) {
      sc[nt] = wmma_f16(a0, bk[nt][0], sc[nt]);
      sc[nt] = wmma_f16(a1, bk[nt][1], sc[nt]);
    }

    // ---- online softmax over this 32-key block (overlaps V-tile loads)
    float corr[8];
#pragma unroll
    for (int v = 0; v < 8; ++v) {
      float s0 = sc[0][v] * ATTN_SCALE;
      float s1 = sc[1][v] * ATTN_SCALE;
      float mx = fmaxf(s0, s1);
      // row reduction over the 16 lanes of this half (masks keep halves apart)
      mx = fmaxf(mx, __shfl_xor(mx, 1, 32));
      mx = fmaxf(mx, __shfl_xor(mx, 2, 32));
      mx = fmaxf(mx, __shfl_xor(mx, 4, 32));
      mx = fmaxf(mx, __shfl_xor(mx, 8, 32));
      float mnew = fmaxf(mi[v], mx);
      float c = __expf(mi[v] - mnew);
      float p0 = __expf(s0 - mnew);
      float p1 = __expf(s1 - mnew);
      float rs = p0 + p1;
      rs += __shfl_xor(rs, 1, 32);
      rs += __shfl_xor(rs, 2, 32);
      rs += __shfl_xor(rs, 4, 32);
      rs += __shfl_xor(rs, 8, 32);
      li[v] = li[v] * c + rs;
      mi[v] = mnew;
      corr[v] = c;
      // P (C layout) -> LDS row-major [16 x 32] f16
      pbuf[wslot][(v + 8 * hi) * 32 + r]      = (_Float16)p0;
      pbuf[wslot][(v + 8 * hi) * 32 + 16 + r] = (_Float16)p1;
    }

    // rescale running accumulator (row of o = v + 8*hi matches corr[v])
#pragma unroll
    for (int dt = 0; dt < 4; ++dt)
#pragma unroll
      for (int v = 0; v < 8; ++v) o[dt][v] *= corr[v];

    // ---- O += P @ V : P as A-operand (16x32), V^T tiles as B-operand
    v16h pa = load_tile_a(&pbuf[wslot][0], 32);
#pragma unroll
    for (int dt = 0; dt < 4; ++dt) o[dt] = wmma_f16(pa, bv[dt], o[dt]);
  }

  // ---- normalize and store attention output f16 [B,S,D]
  _Float16* ob = aoh + ((size_t)(b * S_) + qb * 16) * D_ + h * HD_;
#pragma unroll
  for (int v = 0; v < 8; ++v) {
    float inv = 1.0f / li[v];
#pragma unroll
    for (int dt = 0; dt < 4; ++dt)
      ob[(size_t)(v + 8 * hi) * D_ + dt * 16 + r] = (_Float16)(o[dt][v] * inv);
  }
}

// ---------------------------------------------------------------- launch
extern "C" void kernel_launch(void* const* d_in, const int* in_sizes, int n_in,
                              void* d_out, int out_size, void* d_ws,
                              size_t ws_size, hipStream_t stream) {
  const float* xq = (const float*)d_in[0];
  const float* xk = (const float*)d_in[1];
  const float* xv = (const float*)d_in[2];
  const float* Wq = (const float*)d_in[3];
  const float* bq = (const float*)d_in[4];
  const float* Aq = (const float*)d_in[5];
  const float* Wk = (const float*)d_in[6];
  const float* bk = (const float*)d_in[7];
  const float* Ak = (const float*)d_in[8];
  const float* Wv = (const float*)d_in[9];
  const float* bv = (const float*)d_in[10];
  const float* Av = (const float*)d_in[11];
  const float* Wo = (const float*)d_in[12];
  const float* bo = (const float*)d_in[13];

  const size_t XH = (size_t)M_ * D_;   // 3,145,728 halves per activation
  const size_t WH = (size_t)D_ * D_;   // 589,824 halves per weight

  _Float16* ws = (_Float16*)d_ws;
  _Float16* xqh = ws;            // f16 inputs
  _Float16* xkh = xqh + XH;
  _Float16* xvh = xkh + XH;
  _Float16* Whq = xvh + XH;      // folded weights (W + 4*A*A^T)
  _Float16* Whk = Whq + WH;
  _Float16* Whv = Whk + WH;
  _Float16* Who = Whv + WH;      // Wo as f16 row-major
  _Float16* qh  = Who + WH;      // Q f16 [B*S, D]
  _Float16* khp = qh + XH;       // K f16 [B*S, D]
  _Float16* vth = khp + XH;      // V^T f16 [B*H*64, S]
  _Float16* aoh = vth + XH;      // attention output f16 [B*S, D]
  (void)ws_size; (void)n_in; (void)in_sizes; (void)out_size;

  const int cvt_threads = 256;
  const int cvt_blocks_x = (int)((XH + cvt_threads - 1) / cvt_threads);
  const int cvt_blocks_w = (int)((WH + cvt_threads - 1) / cvt_threads);

  k_f32_to_f16<<<cvt_blocks_x, cvt_threads, 0, stream>>>(xq, xqh, (int)XH);
  k_f32_to_f16<<<cvt_blocks_x, cvt_threads, 0, stream>>>(xk, xkh, (int)XH);
  k_f32_to_f16<<<cvt_blocks_x, cvt_threads, 0, stream>>>(xv, xvh, (int)XH);

  k_build_weight<<<cvt_blocks_w, cvt_threads, 0, stream>>>(Wq, Aq, Whq);
  k_build_weight<<<cvt_blocks_w, cvt_threads, 0, stream>>>(Wk, Ak, Whk);
  k_build_weight<<<cvt_blocks_w, cvt_threads, 0, stream>>>(Wv, Av, Whv);
  k_f32_to_f16<<<cvt_blocks_w, cvt_threads, 0, stream>>>(Wo, Who, (int)WH);

  // GEMM grid: (4096/32) * (768/64) = 1536 waves = 384 blocks x 4 waves
  const int gemm_blocks = (M_ / 32) * (D_ / 64) / 4;  // 384
  k_gemm<<<gemm_blocks, 128, 0, stream>>>(xqh, Whq, bq, qh,  nullptr, 0);
  k_gemm<<<gemm_blocks, 128, 0, stream>>>(xkh, Whk, bk, khp, nullptr, 0);
  k_gemm<<<gemm_blocks, 128, 0, stream>>>(xvh, Whv, bv, vth, nullptr, 1);

  // attention: B*H*(S/16) = 3072 waves = 768 blocks x 4 waves
  const int attn_blocks = (B_ * H_ * (S_ / 16)) / 4;  // 768
  k_attn<<<attn_blocks, 128, 0, stream>>>(qh, khp, vth, aoh);

  // output projection -> f32 d_out
  k_gemm<<<gemm_blocks, 128, 0, stream>>>(aoh, Who, bo, nullptr,
                                          (float*)d_out, 2);
}
